// SAGE_LLPE_31860067402278
// MI455X (gfx1250) — matrix operand order repaired and checked by reading, not compiled, not run on hardware
//
#include <hip/hip_runtime.h>
#include <hip/hip_bf16.h>
#include <cstdint>

typedef __bf16 bf16;
typedef __attribute__((ext_vector_type(16))) __bf16 v16bf;
typedef __attribute__((ext_vector_type(8)))  __bf16 v8bf;
typedef __attribute__((ext_vector_type(8)))  float   v8f;

static constexpr int       NN      = 50000;
static constexpr int       IN_DIM  = 256;
static constexpr int       H_DIM   = 96;
static constexpr int       POS_DIM = 32;
static constexpr int       KCH     = 64;
static constexpr int       MM      = 128;
static constexpr long long EE      = 800000;
static constexpr int       OUT_DIM = 40;
static constexpr int       HID     = 128;

// ---------- float -> bf16 (round to nearest even), packed-pair store ----------
__device__ inline unsigned short f2bf(float f) {
    unsigned u = __float_as_uint(f);
    unsigned r = u + 0x7FFFu + ((u >> 16) & 1u);
    return (unsigned short)(r >> 16);
}
__device__ inline unsigned pack2(float lo, float hi) {
    return (unsigned)f2bf(lo) | ((unsigned)f2bf(hi) << 16);
}

// ---------- monotonic float<->uint encoding for atomic min/max ----------
__device__ inline unsigned encf(float f) {
    unsigned u = __float_as_uint(f);
    return (u & 0x80000000u) ? ~u : (u | 0x80000000u);
}
__device__ inline float decf(unsigned e) {
    unsigned u = (e & 0x80000000u) ? (e ^ 0x80000000u) : ~e;
    return __uint_as_float(u);
}

// ---------- WMMA fragment helpers (bf16 16x16x32, wave32) ----------
// A: 16x32 (MxK) row-major bf16 tile in LDS, row stride ld (elements).
// Lane L<16: M=L, K in {0..7}U{16..23}; lane L>=16: M=L-16, K in {8..15}U{24..31}.
__device__ inline v16bf load_A(const bf16* base, int ld) {
    int lane = threadIdx.x & 31;
    const bf16* row = base + (lane & 15) * ld + ((lane & 16) ? 8 : 0);
    v8bf lo = *(const v8bf*)row;          // K = kh .. kh+7
    v8bf hi = *(const v8bf*)(row + 16);   // K = 16+kh .. 16+kh+7
    return __builtin_shufflevector(lo, hi, 0, 1, 2, 3, 4, 5, 6, 7,
                                   8, 9, 10, 11, 12, 13, 14, 15);
}

// B fragments are pre-swizzled in LDS: frag-ordered [kk][ct][lane][16 elems],
// so a lane's whole fragment is one contiguous 32-byte read (2x ds_load_b128).
__device__ inline v16bf load_B_sw(const bf16* wbuf, int nct, int kk, int ct) {
    int lane = threadIdx.x & 31;
    const bf16* p = wbuf + (size_t)(((kk * nct + ct) * 32) + lane) * 16;
    return *(const v16bf*)p;
}

// Stage a K x (nct*16) B-matrix into fragment order (bf16), packed dword writes.
// trans=true : source S is W[o][k] row-major with row length Kdim (B = W^T)
// trans=false: source S is already B[k][n] row-major with row length ncols_real
__device__ inline void stage_B(bf16* dst, const float* S, int Kdim, int nct,
                               int ncols_real, bool trans, int tid) {
    int total_dw = (Kdim * nct * 16) >> 1;
    unsigned* d32 = (unsigned*)dst;
    for (int j = tid; j < total_dw; j += 256) {
        int i    = j << 1;
        int e    = i & 15;             // even
        int lane = (i >> 4) & 31;
        int t    = i >> 9;             // kk*nct + ct
        int ct   = t % nct;
        int kk   = t / nct;
        int n    = ct * 16 + (lane & 15);
        int k    = kk * 32 + ((lane & 16) ? 8 : 0) + (e & 7) + ((e & 8) ? 16 : 0);
        float v0 = 0.f, v1 = 0.f;
        if (n < ncols_real) {
            if (trans) {
                const float* row = S + (size_t)n * Kdim + k;
                v0 = row[0];
                v1 = row[1];
            } else {
                v0 = S[(size_t)k * ncols_real + n];
                v1 = S[(size_t)(k + 1) * ncols_real + n];
            }
        }
        d32[j] = pack2(v0, v1);
    }
}

__device__ inline v8f wmma_bf16(v16bf a, v16bf b, v8f c) {
    return __builtin_amdgcn_wmma_f32_16x16x32_bf16(false, a, false, b, (short)0, c,
                                                   false, false);
}

// ---------- utility kernels ----------
__global__ void fill_f32(float* p, float v, long long n) {
    long long i      = (long long)blockIdx.x * blockDim.x + threadIdx.x;
    long long stride = (long long)gridDim.x * blockDim.x;
    for (; i < n; i += stride) p[i] = v;
}

__global__ void init_minmax_kernel(unsigned* cmin, unsigned* cmax) {
    int i = threadIdx.x;
    if (i < MM) { cmin[i] = 0xFFFFFFFFu; cmax[i] = 0u; }
}

// one block of 128 threads per 128-row chunk; thread t owns column t
__global__ void ev_minmax_kernel(const float* ev, unsigned* cmin, unsigned* cmax) {
    int col  = threadIdx.x;
    int r0   = blockIdx.x * 128;
    int rend = min(r0 + 128, NN);
    unsigned emin = 0xFFFFFFFFu, emax = 0u;
    for (int r = r0; r < rend; ++r) {
        unsigned e = encf(ev[(size_t)r * MM + col]);
        emin = min(emin, e);
        emax = max(emax, e);
    }
    atomicMin(&cmin[col], emin);
    atomicMax(&cmax[col], emax);
}

// single block: eigenvalue min/max, Chebyshev pe = cos(arccos(lam)*k) @ alpha
__global__ void pe_kernel(const float* eigenvalues, const float* alpha,
                          float* pe_ws, float* out_pe) {
    __shared__ float red[256];
    __shared__ float s_lmin, s_lmax;
    int tid = threadIdx.x;

    red[tid] = (tid < MM) ? eigenvalues[tid] : 3.0e38f;
    __syncthreads();
    for (int s = 128; s; s >>= 1) {
        if (tid < s) red[tid] = fminf(red[tid], red[tid + s]);
        __syncthreads();
    }
    if (tid == 0) s_lmin = red[0];
    __syncthreads();
    red[tid] = (tid < MM) ? eigenvalues[tid] : -3.0e38f;
    __syncthreads();
    for (int s = 128; s; s >>= 1) {
        if (tid < s) red[tid] = fmaxf(red[tid], red[tid + s]);
        __syncthreads();
    }
    if (tid == 0) s_lmax = red[0];
    __syncthreads();

    float lmin = s_lmin, lmax = s_lmax;
    float sc = 2.0f / (lmax - lmin);
    for (int idx = tid; idx < MM * POS_DIM; idx += 256) {
        int m = idx >> 5, p = idx & 31;
        float lam = (eigenvalues[m] - lmin) * sc - 1.0f;
        lam = fminf(1.0f, fmaxf(-1.0f, lam));
        float th  = acosf(lam);
        float acc = 0.f;
#pragma unroll 4
        for (int k = 0; k < KCH; ++k)
            acc += cosf(th * (float)k) * alpha[k * POS_DIM + p];
        pe_ws[idx]  = acc;
        out_pe[idx] = acc;
    }
}

// h0[:, :96] = x @ W_feat.T + b_feat ; h0[:, 96:128] = ev_norm @ pe
__global__ void __launch_bounds__(256)
h0_kernel(const float* x, const float* ev, const unsigned* cmin_e,
          const unsigned* cmax_e, const float* Wf, const float* bfeat,
          const float* pe, float* h0) {
    __shared__ __align__(32) bf16 smem[32 * IN_DIM + IN_DIM * H_DIM];  // 64 KB
    bf16* xt  = smem;                // [32][256] row-major
    bf16* wfB = smem + 32 * IN_DIM;  // frag-ordered W_feat^T (256 x 96)
    int tid   = threadIdx.x;
    int node0 = blockIdx.x * 32;

    // ---- phase A: feature GEMM (K = 256) ----
    {
        unsigned* d32 = (unsigned*)xt;
        for (int j = tid; j < (32 * IN_DIM) >> 1; j += 256) {
            int i = j << 1;
            int r = i >> 8, c = i & 255;
            int node = node0 + r;
            float v0 = 0.f, v1 = 0.f;
            if (node < NN) {
                const float* p = x + (size_t)node * IN_DIM + c;
                v0 = p[0];
                v1 = p[1];
            }
            d32[j] = pack2(v0, v1);
        }
    }
    stage_B(wfB, Wf, IN_DIM, 6, H_DIM, true, tid);
    __syncthreads();

    int w   = tid >> 5;
    int t0  = 2 * w, t1 = 2 * w + 1;
    int rt0 = t0 >> 3, ct0 = t0 & 7;
    int rt1 = t1 >> 3, ct1 = t1 & 7;
    v8f c0 = {}, c1 = {};

    if (ct0 < 6) {
#pragma unroll
        for (int kk = 0; kk < IN_DIM / 32; ++kk) {
            v16bf a = load_A(xt + rt0 * 16 * IN_DIM + kk * 32, IN_DIM);
            v16bf b = load_B_sw(wfB, 6, kk, ct0);
            c0 = wmma_bf16(a, b, c0);
        }
    }
    if (ct1 < 6) {
#pragma unroll
        for (int kk = 0; kk < IN_DIM / 32; ++kk) {
            v16bf a = load_A(xt + rt1 * 16 * IN_DIM + kk * 32, IN_DIM);
            v16bf b = load_B_sw(wfB, 6, kk, ct1);
            c1 = wmma_bf16(a, b, c1);
        }
    }
    __syncthreads();

    // ---- phase B: positional GEMM (K = 128), reuse LDS ----
    bf16* evt = smem;             // [32][128] normalized eigenvectors
    bf16* peB = smem + 32 * HID;  // frag-ordered pe (128 x 32)
    {
        unsigned* d32 = (unsigned*)evt;
        for (int j = tid; j < (32 * HID) >> 1; j += 256) {
            int i = j << 1;
            int r = i >> 7, c = i & 127;
            int node = node0 + r;
            float v0 = 0.f, v1 = 0.f;
            if (node < NN) {
                float mn0 = decf(cmin_e[c]), mx0 = decf(cmax_e[c]);
                float mn1 = decf(cmin_e[c + 1]), mx1 = decf(cmax_e[c + 1]);
                const float* p = ev + (size_t)node * MM + c;
                v0 = (p[0] - mn0) * (2.0f / (mx0 - mn0)) - 1.0f;
                v1 = (p[1] - mn1) * (2.0f / (mx1 - mn1)) - 1.0f;
            }
            d32[j] = pack2(v0, v1);
        }
    }
    stage_B(peB, pe, HID, 2, POS_DIM, false, tid);
    __syncthreads();

    if (ct0 >= 6) {
#pragma unroll
        for (int kk = 0; kk < HID / 32; ++kk) {
            v16bf a = load_A(evt + rt0 * 16 * HID + kk * 32, HID);
            v16bf b = load_B_sw(peB, 2, kk, ct0 - 6);
            c0 = wmma_bf16(a, b, c0);
        }
    }
    if (ct1 >= 6) {
#pragma unroll
        for (int kk = 0; kk < HID / 32; ++kk) {
            v16bf a = load_A(evt + rt1 * 16 * HID + kk * 32, HID);
            v16bf b = load_B_sw(peB, 2, kk, ct1 - 6);
            c1 = wmma_bf16(a, b, c1);
        }
    }

    // ---- epilogue ----
    int lane = tid & 31;
    int n = lane & 15;
    int mb = (lane & 16) ? 8 : 0;
    {
        int col    = (ct0 < 6) ? (ct0 * 16 + n) : (H_DIM + (ct0 - 6) * 16 + n);
        float bias = (ct0 < 6) ? bfeat[ct0 * 16 + n] : 0.f;
#pragma unroll
        for (int r = 0; r < 8; ++r) {
            int node = node0 + rt0 * 16 + mb + r;
            if (node < NN) h0[(size_t)node * HID + col] = c0[r] + bias;
        }
    }
    {
        int col    = (ct1 < 6) ? (ct1 * 16 + n) : (H_DIM + (ct1 - 6) * 16 + n);
        float bias = (ct1 < 6) ? bfeat[ct1 * 16 + n] : 0.f;
#pragma unroll
        for (int r = 0; r < 8; ++r) {
            int node = node0 + rt1 * 16 + mb + r;
            if (node < NN) h0[(size_t)node * HID + col] = c1[r] + bias;
        }
    }
}

// one 32-lane wave per edge: lane moves one float4, L2-resident atomics
__global__ void scatter_kernel(const float* __restrict__ h, const int* __restrict__ src,
                               const int* __restrict__ dst, float* agg, float* cnt) {
    long long tidg = (long long)blockIdx.x * blockDim.x + threadIdx.x;
    long long e    = tidg >> 5;
    if (e >= EE) return;
    int lane = (int)(tidg & 31);
    int s = src[e], d = dst[e];
    const float4* hs = (const float4*)(h + (size_t)s * HID);
    float4 v  = hs[lane];
    float* ad = agg + (size_t)d * HID + lane * 4;
    atomicAdd(ad + 0, v.x);
    atomicAdd(ad + 1, v.y);
    atomicAdd(ad + 2, v.z);
    atomicAdd(ad + 3, v.w);
    if (lane == 0) atomicAdd(&cnt[d], 1.0f);
}

// out = relu(mean @ Wl.T + bl + h @ Wr.T)   (both GEMMs share one accumulator)
__global__ void __launch_bounds__(256)
sage_kernel(const float* hin, const float* agg, const float* cnt, const float* Wl,
            const float* bl, const float* Wr, float* hout) {
    __shared__ __align__(32) bf16 smem[32 * HID + HID * HID];  // 40 KB
    bf16* at = smem;             // [32][128] (mean, then h) row-major
    bf16* wB = smem + 32 * HID;  // frag-ordered weight (128 x 128)
    int tid   = threadIdx.x;
    int node0 = blockIdx.x * 32;

    int w   = tid >> 5;
    int t0  = 2 * w, t1 = 2 * w + 1;
    int rt0 = t0 >> 3, ct0 = t0 & 7;
    int rt1 = t1 >> 3, ct1 = t1 & 7;
    v8f c0 = {}, c1 = {};

    // phase 1: mean x Wl^T
    {
        unsigned* d32 = (unsigned*)at;
        for (int j = tid; j < (32 * HID) >> 1; j += 256) {
            int i = j << 1;
            int r = i >> 7, c = i & 127;
            int node = node0 + r;
            float v0 = 0.f, v1 = 0.f;
            if (node < NN) {
                float rc = 1.0f / fmaxf(cnt[node], 1.0f);
                const float* p = agg + (size_t)node * HID + c;
                v0 = p[0] * rc;
                v1 = p[1] * rc;
            }
            d32[j] = pack2(v0, v1);
        }
    }
    stage_B(wB, Wl, HID, 8, HID, true, tid);
    __syncthreads();
#pragma unroll
    for (int kk = 0; kk < HID / 32; ++kk) {
        v16bf a0 = load_A(at + rt0 * 16 * HID + kk * 32, HID);
        c0 = wmma_bf16(a0, load_B_sw(wB, 8, kk, ct0), c0);
        v16bf a1 = load_A(at + rt1 * 16 * HID + kk * 32, HID);
        c1 = wmma_bf16(a1, load_B_sw(wB, 8, kk, ct1), c1);
    }
    __syncthreads();

    // phase 2: h x Wr^T (same accumulators)
    {
        unsigned* d32 = (unsigned*)at;
        for (int j = tid; j < (32 * HID) >> 1; j += 256) {
            int i = j << 1;
            int r = i >> 7, c = i & 127;
            int node = node0 + r;
            float v0 = 0.f, v1 = 0.f;
            if (node < NN) {
                const float* p = hin + (size_t)node * HID + c;
                v0 = p[0];
                v1 = p[1];
            }
            d32[j] = pack2(v0, v1);
        }
    }
    stage_B(wB, Wr, HID, 8, HID, true, tid);
    __syncthreads();
#pragma unroll
    for (int kk = 0; kk < HID / 32; ++kk) {
        v16bf a0 = load_A(at + rt0 * 16 * HID + kk * 32, HID);
        c0 = wmma_bf16(a0, load_B_sw(wB, 8, kk, ct0), c0);
        v16bf a1 = load_A(at + rt1 * 16 * HID + kk * 32, HID);
        c1 = wmma_bf16(a1, load_B_sw(wB, 8, kk, ct1), c1);
    }

    int lane = tid & 31;
    int n = lane & 15;
    int mb = (lane & 16) ? 8 : 0;
    {
        float bias = bl[ct0 * 16 + n];
#pragma unroll
        for (int r = 0; r < 8; ++r) {
            int node = node0 + rt0 * 16 + mb + r;
            if (node < NN)
                hout[(size_t)node * HID + ct0 * 16 + n] = fmaxf(c0[r] + bias, 0.f);
        }
    }
    {
        float bias = bl[ct1 * 16 + n];
#pragma unroll
        for (int r = 0; r < 8; ++r) {
            int node = node0 + rt1 * 16 + mb + r;
            if (node < NN)
                hout[(size_t)node * HID + ct1 * 16 + n] = fmaxf(c1[r] + bias, 0.f);
        }
    }
}

// logits = h @ W_ro.T + b_ro   (40 cols padded to 48 -> 3 col-tiles)
__global__ void __launch_bounds__(256)
logits_kernel(const float* h, const float* Wro, const float* bro, float* out) {
    __shared__ __align__(32) bf16 smem[32 * HID + HID * 48];  // 20 KB
    bf16* ht = smem;             // [32][128]
    bf16* wB = smem + 32 * HID;  // frag-ordered W_ro^T (128 x 48, cols 40..47 zero)
    int tid   = threadIdx.x;
    int node0 = blockIdx.x * 32;

    {
        unsigned* d32 = (unsigned*)ht;
        for (int j = tid; j < (32 * HID) >> 1; j += 256) {
            int i = j << 1;
            int r = i >> 7, c = i & 127;
            int node = node0 + r;
            float v0 = 0.f, v1 = 0.f;
            if (node < NN) {
                const float* p = h + (size_t)node * HID + c;
                v0 = p[0];
                v1 = p[1];
            }
            d32[j] = pack2(v0, v1);
        }
    }
    stage_B(wB, Wro, HID, 3, OUT_DIM, true, tid);
    __syncthreads();

    int w = tid >> 5;
    if (w < 6) {
        int rt = w / 3, ct = w % 3;
        v8f c = {};
#pragma unroll
        for (int kk = 0; kk < HID / 32; ++kk) {
            v16bf a = load_A(ht + rt * 16 * HID + kk * 32, HID);
            c = wmma_bf16(a, load_B_sw(wB, 3, kk, ct), c);
        }
        int lane = tid & 31;
        int n = lane & 15;
        int mb = (lane & 16) ? 8 : 0;
        int col = ct * 16 + n;
        if (col < OUT_DIM) {
            float bias = bro[col];
#pragma unroll
            for (int r = 0; r < 8; ++r) {
                int node = node0 + rt * 16 + mb + r;
                if (node < NN) out[(size_t)node * OUT_DIM + col] = c[r] + bias;
            }
        }
    }
}

__global__ void logsoftmax_kernel(float* out) {
    int n = blockIdx.x * blockDim.x + threadIdx.x;
    if (n >= NN) return;
    float* row = out + (size_t)n * OUT_DIM;
    float m = -3.0e38f;
#pragma unroll
    for (int i = 0; i < OUT_DIM; ++i) m = fmaxf(m, row[i]);
    float s = 0.f;
#pragma unroll
    for (int i = 0; i < OUT_DIM; ++i) s += expf(row[i] - m);
    float ls = logf(s) + m;
#pragma unroll
    for (int i = 0; i < OUT_DIM; ++i) row[i] -= ls;
}

extern "C" void kernel_launch(void* const* d_in, const int* in_sizes, int n_in,
                              void* d_out, int out_size, void* d_ws, size_t ws_size,
                              hipStream_t stream) {
    (void)in_sizes; (void)n_in; (void)out_size; (void)ws_size;
    const float* x     = (const float*)d_in[0];
    const float* ev    = (const float*)d_in[1];
    const float* evals = (const float*)d_in[2];
    const int*   eidx  = (const int*)d_in[3];   // (2, E) int32
    const float* Wf    = (const float*)d_in[4];
    const float* bfeat = (const float*)d_in[5];
    const float* alpha = (const float*)d_in[6];
    const float* Wl1   = (const float*)d_in[7];
    const float* bl1   = (const float*)d_in[8];
    const float* Wr1   = (const float*)d_in[9];
    const float* Wl2   = (const float*)d_in[10];
    const float* bl2   = (const float*)d_in[11];
    const float* Wr2   = (const float*)d_in[12];
    const float* Wro   = (const float*)d_in[13];
    const float* bro   = (const float*)d_in[14];
    float* out = (float*)d_out;

    // workspace layout (floats)
    float*    ws     = (float*)d_ws;
    unsigned* cmin_e = (unsigned*)ws;          // 128
    unsigned* cmax_e = (unsigned*)(ws + 128);  // 128
    float* pe  = ws + 256;                     // 4096
    float* hA  = ws + 4352;                    // N*HID
    float* hB  = hA + (size_t)NN * HID;        // N*HID
    float* cnt = hB + (size_t)NN * HID;        // N  (contiguous with agg)
    float* agg = cnt + NN;                     // N*HID

    const int* srcI = eidx;
    const int* dstI = eidx + EE;

    const int nb32    = (NN + 31) / 32;                // 1563
    const int sblocks = (int)((EE * 32 + 255) / 256);  // 100000
    const long long zcount = (long long)NN + (long long)NN * HID;

    init_minmax_kernel<<<1, 128, 0, stream>>>(cmin_e, cmax_e);
    ev_minmax_kernel<<<(NN + 127) / 128, 128, 0, stream>>>(ev, cmin_e, cmax_e);
    pe_kernel<<<1, 256, 0, stream>>>(evals, alpha, pe, out + (size_t)NN * OUT_DIM);
    h0_kernel<<<nb32, 256, 0, stream>>>(x, ev, cmin_e, cmax_e, Wf, bfeat, pe, hA);

    // layer 1
    fill_f32<<<2048, 256, 0, stream>>>(cnt, 0.f, zcount);
    scatter_kernel<<<sblocks, 256, 0, stream>>>(hA, srcI, dstI, agg, cnt);
    sage_kernel<<<nb32, 256, 0, stream>>>(hA, agg, cnt, Wl1, bl1, Wr1, hB);

    // layer 2
    fill_f32<<<2048, 256, 0, stream>>>(cnt, 0.f, zcount);
    scatter_kernel<<<sblocks, 256, 0, stream>>>(hB, srcI, dstI, agg, cnt);
    sage_kernel<<<nb32, 256, 0, stream>>>(hB, agg, cnt, Wl2, bl2, Wr2, hA);

    // readout
    logits_kernel<<<nb32, 256, 0, stream>>>(hA, Wro, bro, out);
    logsoftmax_kernel<<<(NN + 255) / 256, 256, 0, stream>>>(out);
}
